// GAT_ml_25m_14362370638272
// MI455X (gfx1250) — matrix operand order repaired and checked by reading, not compiled, not run on hardware
//
#include <hip/hip_runtime.h>
#include <hip/hip_bf16.h>

typedef __attribute__((ext_vector_type(16))) _Float16 v16h;
typedef __attribute__((ext_vector_type(8)))  float    v8f;

// ---------------------------------------------------------------------------
// Pack W[K,M] (f32 row-major) into f16 B-fragment order:
// fragment t = kc*tilesM + ct covers rows kc*32..+31, cols ct*16..+15.
// Element (t, lane, j):  h=lane>>4, mr=lane&15, v=j>>1, lo=j&1
//   -> W[kc*32 + (j>>3)*16 + h*8 + (j&7)][ct*16 + mr]
// Each lane of proj_wmma then loads its whole fragment as one aligned v16h.
// ---------------------------------------------------------------------------
__global__ void pack_w(const float* __restrict__ W, _Float16* __restrict__ Wp,
                       int K, int M) {
  int i = blockIdx.x * blockDim.x + threadIdx.x;
  if (i >= K * M) return;
  int tilesM = M >> 4;
  int j    = i & 15;
  int lane = (i >> 4) & 31;
  int t    = i >> 9;
  int ct = t % tilesM, kc = t / tilesM;
  int h = lane >> 4, mr = lane & 15;
  int v = j >> 1, lo = j & 1;
  int kr = kc * 32 + (j >> 3) * 16 + h * 8 + (j & 7);
  Wp[i] = (_Float16)W[(size_t)kr * M + ct * 16 + mr];
}

// ---------------------------------------------------------------------------
// out[N,M] = X[N,K] @ W[K,M] (+ bias).  f32 in/out, f16 WMMA math.
// One wave computes a 16x32 output block (two 16x16 tiles, two accumulators).
// REQUIRES: N % 16 == 0, K % 32 == 0, M % 32 == 0 (true for all uses here).
// ---------------------------------------------------------------------------
__global__ __launch_bounds__(256)
void proj_wmma(const float* __restrict__ X, const _Float16* __restrict__ Wp,
               const float* __restrict__ bias, float* __restrict__ out,
               int N, int K, int M) {
  int wave   = (blockIdx.x * blockDim.x + threadIdx.x) >> 5;
  int lane   = threadIdx.x & 31;
  int pairsM = M >> 5;
  int rowT   = wave / pairsM;
  int cp     = wave % pairsM;
  if (rowT >= (N >> 4)) return;               // wave-uniform exit
  int row0 = rowT << 4, col0 = cp << 5;
  int h  = lane >> 4;                         // lane half (0/1)
  int mr = lane & 15;                         // A row / B-C column within tile
  int tilesM = M >> 4;

  const float* xr = X + (size_t)(row0 + mr) * K;
  const v16h*  Wv = (const v16h*)Wp;

  v8f acc0 = {}, acc1 = {};
  for (int k0 = 0, kc = 0; k0 < K; k0 += 32, ++kc) {
    // A fragment: a[0..7] = X[row][k0 + h*8 + 0..7], a[8..15] = X[row][k0+16+h*8+0..7]
    float4 a0 = *(const float4*)(xr + k0 + h * 8);
    float4 a1 = *(const float4*)(xr + k0 + h * 8 + 4);
    float4 a2 = *(const float4*)(xr + k0 + 16 + h * 8);
    float4 a3 = *(const float4*)(xr + k0 + 16 + h * 8 + 4);
    v16h a;
    a[0]=(_Float16)a0.x;  a[1]=(_Float16)a0.y;  a[2]=(_Float16)a0.z;  a[3]=(_Float16)a0.w;
    a[4]=(_Float16)a1.x;  a[5]=(_Float16)a1.y;  a[6]=(_Float16)a1.z;  a[7]=(_Float16)a1.w;
    a[8]=(_Float16)a2.x;  a[9]=(_Float16)a2.y;  a[10]=(_Float16)a2.z; a[11]=(_Float16)a2.w;
    a[12]=(_Float16)a3.x; a[13]=(_Float16)a3.y; a[14]=(_Float16)a3.z; a[15]=(_Float16)a3.w;

    int t0 = kc * tilesM + 2 * cp;
    v16h b0 = Wv[(size_t)t0 * 32 + lane];
    v16h b1 = Wv[(size_t)(t0 + 1) * 32 + lane];

    acc0 = __builtin_amdgcn_wmma_f32_16x16x32_f16(false, a, false, b0,
                                                  (short)0, acc0, false, false);
    acc1 = __builtin_amdgcn_wmma_f32_16x16x32_f16(false, a, false, b1,
                                                  (short)0, acc1, false, false);
  }
  // C/D layout: vgpr v holds row (v + 8*h), column mr
  float bv0 = bias ? bias[col0 + mr]      : 0.f;
  float bv1 = bias ? bias[col0 + 16 + mr] : 0.f;
  #pragma unroll
  for (int v = 0; v < 8; ++v) {
    int r = row0 + v + 8 * h;
    out[(size_t)r * M + col0 + mr]      = acc0[v] + bv0;
    out[(size_t)r * M + col0 + 16 + mr] = acc1[v] + bv1;
  }
}

// ---------------------------------------------------------------------------
// Front-end helpers
// ---------------------------------------------------------------------------
__global__ void fill_zero(float* p, long long n) {
  long long i = (long long)blockIdx.x * blockDim.x + threadIdx.x;
  if (i < n) p[i] = 0.f;
}

__global__ void gather_emb(const float* __restrict__ tab, const int* __restrict__ ids,
                           float* __restrict__ out, int n) {
  int i = blockIdx.x * blockDim.x + threadIdx.x;
  if (i >= n * 32) return;
  int node = i >> 5, d = i & 31;
  out[i] = tab[(size_t)ids[node] * 32 + d];
}

__global__ void build_item_x(const float* __restrict__ tab, const int* __restrict__ id,
                             const int* __restrict__ rating, const float* __restrict__ ts,
                             float* __restrict__ out, int n) {
  int i = blockIdx.x * blockDim.x + threadIdx.x;
  if (i >= n * 32) return;
  int node = i >> 5, d = i & 31;
  float* o = out + (size_t)node * 96;
  o[d]      = tab[(size_t)id[node]     * 32 + d];
  o[32 + d] = tab[(size_t)rating[node] * 32 + d];
  o[64 + d] = ts[i];
}

// wl[k,h] = sum_d W[k, h*D+d]*al[h,d]   (and wr with ar) -> el/er become x @ wl
__global__ void compute_wlr(const float* __restrict__ W, const float* __restrict__ al,
                            const float* __restrict__ ar, float* __restrict__ wl,
                            float* __restrict__ wr, int K, int H, int D) {
  int i = blockIdx.x * blockDim.x + threadIdx.x;
  if (i >= K * H) return;
  int k = i / H, h = i % H;
  const float* wrow = W + (size_t)k * H * D + h * D;
  float sl = 0.f, sr = 0.f;
  for (int d = 0; d < D; ++d) { sl += wrow[d] * al[h * D + d]; sr += wrow[d] * ar[h * D + d]; }
  wl[i] = sl; wr[i] = sr;
}

__global__ void node_scores(const float* __restrict__ x, const float* __restrict__ wm,
                            float* __restrict__ s, int N, int K, int H) {
  int i = blockIdx.x * blockDim.x + threadIdx.x;
  if (i >= N * H) return;
  int n = i / H, h = i % H;
  const float* xr = x + (size_t)n * K;
  float acc = 0.f;
  for (int k = 0; k < K; ++k) acc += xr[k] * wm[k * H + h];
  s[i] = acc;
}

// ---------------------------------------------------------------------------
// Edge softmax + aggregation (ordered-uint float atomicMax trick)
// ---------------------------------------------------------------------------
__device__ __forceinline__ unsigned f2ord(float f) {
  unsigned u = __float_as_uint(f);
  return (u & 0x80000000u) ? ~u : (u | 0x80000000u);
}
__device__ __forceinline__ float ord2f(unsigned u) {
  return __uint_as_float((u & 0x80000000u) ? (u & 0x7FFFFFFFu) : ~u);
}

__global__ void edge_scores(const int* __restrict__ src, const int* __restrict__ dst,
                            const float* __restrict__ el, const float* __restrict__ er,
                            float* __restrict__ e, unsigned* __restrict__ emax, int E, int H) {
  int i = blockIdx.x * blockDim.x + threadIdx.x;
  if (i >= E * H) return;
  int edge = i / H, h = i % H;
  float v = el[src[edge] * H + h] + er[dst[edge] * H + h];
  v = v > 0.f ? v : 0.2f * v;                 // leaky_relu(0.2)
  e[i] = v;
  atomicMax(&emax[dst[edge] * H + h], f2ord(v));
}

__global__ void edge_exp(const int* __restrict__ dst, float* __restrict__ e,
                         const unsigned* __restrict__ emax, float* __restrict__ denom,
                         int E, int H) {
  int i = blockIdx.x * blockDim.x + threadIdx.x;
  if (i >= E * H) return;
  int edge = i / H, h = i % H;
  float ex = __expf(e[i] - ord2f(emax[dst[edge] * H + h]));
  e[i] = ex;
  atomicAdd(&denom[dst[edge] * H + h], ex);
}

// out[dst] += fs[src, h*D+d] * (ee/denom).  D lanes per edge, H elems per lane.
__global__ void edge_aggregate(const int* __restrict__ src, const int* __restrict__ dst,
                               const float* __restrict__ ee, const float* __restrict__ denom,
                               const float* __restrict__ fs, float* __restrict__ out,
                               int E, int H, int D) {
  long long t = (long long)blockIdx.x * blockDim.x + threadIdx.x;
  if (t >= (long long)E * D) return;
  int edge = (int)(t / D), d = (int)(t % D);
  int s = src[edge], q = dst[edge];
  const float* fr = fs + (size_t)s * H * D;
  float* orow     = out + (size_t)q * H * D;
  __builtin_prefetch(fr, 0, 1);               // global_prefetch_b8
  for (int h = 0; h < H; ++h) {
    float alpha = ee[(size_t)edge * H + h] / denom[(size_t)q * H + h];
    atomicAdd(&orow[h * D + d], fr[h * D + d] * alpha);
  }
}

__global__ void add_bias(float* __restrict__ out, const float* __restrict__ b, int N, int F) {
  long long i = (long long)blockIdx.x * blockDim.x + threadIdx.x;
  if (i >= (long long)N * F) return;
  out[i] += b[i % F];
}

__global__ void final_cos(const float* __restrict__ h2_item, const float* __restrict__ h2_user,
                          const int* __restrict__ target, float* __restrict__ out, int bs) {
  int n = blockIdx.x * blockDim.x + threadIdx.x;
  if (n >= bs) return;
  const float* t = h2_item + (size_t)target[n] * 32;
  const float* u = h2_user + (size_t)n * 32;
  float uu = 0.f, tt = 0.f, ut = 0.f;
  for (int d = 0; d < 32; ++d) { float a = u[d], b = t[d]; uu += a * a; tt += b * b; ut += a * b; }
  float un = fmaxf(sqrtf(uu), 1e-8f), tn = fmaxf(sqrtf(tt), 1e-8f);
  float c = ut / (un * tn);
  out[n] = 1.f / (1.f + __expf(-c));
}

// ---------------------------------------------------------------------------
static inline int nblk(long long n, int b) { return (int)((n + b - 1) / b); }

extern "C" void kernel_launch(void* const* d_in, const int* in_sizes, int n_in,
                              void* d_out, int out_size, void* d_ws, size_t ws_size,
                              hipStream_t stream) {
  (void)n_in; (void)out_size; (void)ws_size;
  const int BS = in_sizes[0];
  const int NI = in_sizes[1];
  const int NF = in_sizes[4];
  const int EB = in_sizes[5] / 2, EF = in_sizes[6] / 2,
            EU = in_sizes[7] / 2, EC = in_sizes[8] / 2;
  const int H1 = 8, D = 32, F1 = H1 * D;      // layer-1: 8 heads x 32

  const int*   user_feat   = (const int*)  d_in[0];
  const int*   item_id     = (const int*)  d_in[1];
  const int*   item_rating = (const int*)  d_in[2];
  const float* item_ts     = (const float*)d_in[3];
  const int*   feat_id     = (const int*)  d_in[4];
  const int*   e_bf        = (const int*)  d_in[5];
  const int*   e_fi        = (const int*)  d_in[6];
  const int*   e_ui        = (const int*)  d_in[7];
  const int*   e_iu        = (const int*)  d_in[8];
  const int*   target_idx  = (const int*)  d_in[9];
  const float* emb_table   = (const float*)d_in[10];
  const float* user_W = (const float*)d_in[11]; const float* user_b = (const float*)d_in[12];
  const float* item_W = (const float*)d_in[13]; const float* item_b = (const float*)d_in[14];
  const float* feat_W = (const float*)d_in[15]; const float* feat_b = (const float*)d_in[16];
  // l1_params: belongto=17..20 hasinstance=21..24 interacted=25..28 clickby=29..32
  // l2_params: belongto=33..36 hasinstance=37..40 interacted=41..44 clickby=45..48

  // ---- workspace (bump allocator, floats; every size is a multiple of 8
  //      floats = 32 bytes, so v16h loads from wpack stay aligned) ----
  size_t off = 0;
  auto alloc = [&](size_t nf) { float* p = (float*)d_ws + off; off += nf; return p; };
  float* user_x   = alloc((size_t)BS * 32);
  float* feat_x   = alloc((size_t)NF * 32);
  float* user_emb = alloc((size_t)BS * 32);
  float* item_emb = alloc((size_t)NI * 32);
  float* feat_emb = alloc((size_t)NF * 32);
  float* h_item   = alloc((size_t)NI * F1);
  float* h_feat   = alloc((size_t)NF * F1);
  float* h_user   = alloc((size_t)BS * F1);
  float* fs       = alloc((size_t)NI * F1);   // also aliased as item_x (NI*96 fits)
  float* el       = alloc((size_t)NI * H1);
  float* er       = alloc((size_t)NI * H1);
  float* ebuf     = alloc((size_t)(EU > EC ? EU : EC) * H1);
  float* emaxb    = alloc((size_t)NI * H1);
  float* denom    = alloc((size_t)NI * H1);
  float* wl       = alloc(256 * 8);
  float* wr       = alloc(256 * 8);
  float* h2_item  = alloc((size_t)NI * 32);
  float* h2_user  = alloc((size_t)BS * 32);
  _Float16* wpack = (_Float16*)alloc(256 * 256 / 2 + 8);  // f16 packed W (max 64K halves)
  float* item_x   = fs;

  auto zero = [&](float* p, long long n) {
    fill_zero<<<nblk(n, 256), 256, 0, stream>>>(p, n);
  };
  // pack W to f16 fragment order, then WMMA projection
  auto proj = [&](const float* X, const float* W, const float* b, float* O,
                  int N, int K, int M) {
    pack_w<<<nblk((long long)K * M, 256), 256, 0, stream>>>(W, wpack, K, M);
    long long waves = (long long)(N / 16) * (M / 32);
    proj_wmma<<<nblk(waves * 32, 256), 256, 0, stream>>>(X, wpack, b, O, N, K, M);
  };

  // One full GATConv relation: fs = xs@W, el/er via reduced matrices, edge softmax,
  // scatter-aggregate into outbuf (+bias).
  auto run_gat = [&](const float* xs, int Ns, const float* xd, int Nd, int K,
                     const int* eptr, int E, int pbase, int H, float* outbuf) {
    const float* W  = (const float*)d_in[pbase];
    const float* al = (const float*)d_in[pbase + 1];
    const float* ar = (const float*)d_in[pbase + 2];
    const float* b  = (const float*)d_in[pbase + 3];
    const int* src = eptr; const int* dst = eptr + E;
    int F = H * D;
    compute_wlr<<<nblk((long long)K * H, 256), 256, 0, stream>>>(W, al, ar, wl, wr, K, H, D);
    proj(xs, W, nullptr, fs, Ns, K, F);
    node_scores<<<nblk((long long)Ns * H, 256), 256, 0, stream>>>(xs, wl, el, Ns, K, H);
    node_scores<<<nblk((long long)Nd * H, 256), 256, 0, stream>>>(xd, wr, er, Nd, K, H);
    zero(emaxb, (long long)Nd * H);           // ordered key 0 == identity for max
    zero(denom, (long long)Nd * H);
    edge_scores<<<nblk((long long)E * H, 256), 256, 0, stream>>>(
        src, dst, el, er, ebuf, (unsigned*)emaxb, E, H);
    edge_exp<<<nblk((long long)E * H, 256), 256, 0, stream>>>(
        dst, ebuf, (const unsigned*)emaxb, denom, E, H);
    edge_aggregate<<<nblk((long long)E * D, 256), 256, 0, stream>>>(
        src, dst, ebuf, denom, fs, outbuf, E, H, D);
    add_bias<<<nblk((long long)Nd * F, 256), 256, 0, stream>>>(outbuf, b, Nd, F);
  };

  // ---- front end: gathers + node-type linears (WMMA) ----
  gather_emb<<<nblk((long long)BS * 32, 256), 256, 0, stream>>>(emb_table, user_feat, user_x, BS);
  gather_emb<<<nblk((long long)NF * 32, 256), 256, 0, stream>>>(emb_table, feat_id, feat_x, NF);
  build_item_x<<<nblk((long long)NI * 32, 256), 256, 0, stream>>>(
      emb_table, item_id, item_rating, item_ts, item_x, NI);
  proj(user_x, user_W, user_b, user_emb, BS, 32, 32);
  proj(item_x, item_W, item_b, item_emb, NI, 96, 32);
  proj(feat_x, feat_W, feat_b, feat_emb, NF, 32, 32);

  // ---- layer 1 (8 heads), HeteroGraphConv sum-aggregate per dst type ----
  zero(h_item, (long long)NI * F1);
  zero(h_feat, (long long)NF * F1);
  zero(h_user, (long long)BS * F1);
  run_gat(item_emb, NI, feat_emb, NF, 32, e_bf, EB, 17, H1, h_feat);   // belongto
  run_gat(feat_emb, NF, item_emb, NI, 32, e_fi, EF, 21, H1, h_item);   // hasinstance
  run_gat(user_emb, BS, item_emb, NI, 32, e_ui, EU, 25, H1, h_item);   // interacted
  run_gat(item_emb, NI, user_emb, BS, 32, e_iu, EC, 29, H1, h_user);   // clickby

  // ---- layer 2 (1 head, din=256) ----
  zero(h2_item, (long long)NI * 32);
  zero(h2_user, (long long)BS * 32);
  run_gat(h_feat, NF, h_item, NI, 256, e_fi, EF, 37, 1, h2_item);      // hasinstance
  run_gat(h_user, BS, h_item, NI, 256, e_ui, EU, 41, 1, h2_item);      // interacted
  run_gat(h_item, NI, h_user, BS, 256, e_iu, EC, 45, 1, h2_user);      // clickby

  // ---- cosine similarity + sigmoid ----
  final_cos<<<nblk(BS, 256), 256, 0, stream>>>(h2_item, h2_user, target_idx,
                                               (float*)d_out, BS);
}
